// WhisperSelfAttentionWithCache_79336635892011
// MI455X (gfx1250) — compile-verified, matrix operand order checked by reading, not run on hardware
//
#include <hip/hip_runtime.h>
#include <stdint.h>

// ---------------------------------------------------------------------------
// WhisperSelfAttentionWithCache for MI455X (gfx1250, wave32, WMMA).
// Pipeline: f32->bf16 convert, QKV GEMMs (v_wmma_f32_16x16x32_bf16) with
// async global->LDS double-buffered staging, flash-attention with online
// softmax, output projection GEMM (f32 out).
// ---------------------------------------------------------------------------

typedef __attribute__((ext_vector_type(16))) __bf16       v16bf;
typedef __attribute__((ext_vector_type(8)))  float        v8f;
typedef __attribute__((ext_vector_type(4)))  unsigned int u32x4;
typedef int gv4i __attribute__((vector_size(16)));   // matches builtin proto

#define EMBED 1280
#define NHEAD 20
#define HDIM  64
#define SEQ   2048
#define NEGINF -1e30f

#define AS1 __attribute__((address_space(1)))
#define AS3 __attribute__((address_space(3)))

#if defined(__has_builtin)
#if __has_builtin(__builtin_amdgcn_global_load_async_to_lds_b128) && \
    __has_builtin(__builtin_amdgcn_s_wait_asynccnt)
#define HAVE_ASYNC_LDS 1
#endif
#endif
#ifndef HAVE_ASYNC_LDS
#define HAVE_ASYNC_LDS 0
#endif

#if HAVE_ASYNC_LDS
#define WAIT_ASYNC(n) __builtin_amdgcn_s_wait_asynccnt(n)
#else
#define WAIT_ASYNC(n)
#endif

// 16B global -> LDS copy: async (ASYNCcnt-tracked, no VGPR round trip) when
// available, otherwise load/store through registers.
__device__ __forceinline__ void copy16_g2l(const unsigned short* g,
                                           unsigned short* l) {
#if HAVE_ASYNC_LDS
  __builtin_amdgcn_global_load_async_to_lds_b128(
      (AS1 gv4i*)const_cast<unsigned short*>(g), (AS3 gv4i*)l, 0, 0);
#else
  *(u32x4*)l = *(const u32x4*)g;
#endif
}

__device__ __forceinline__ unsigned short f2bf(float x) {
  unsigned u = __float_as_uint(x);
  unsigned r = u + 0x7FFFu + ((u >> 16) & 1u);   // round-to-nearest-even
  return (unsigned short)(r >> 16);
}

// A/B fragment for v_wmma_*_16x16x32_bf16: 16 bf16 = two 16B chunks.
// Per ISA 16-bit A layout: lane&15 = row, (lane>>4)*8 = K-chunk base,
// chunk0 = K[kb..kb+7], chunk1 = K[kb+16..kb+23].
union AFrag { v16bf bf; u32x4 u[2]; };

// ---------------------------------------------------------------------------
// fp32 -> bf16 (vectorized, n4 = n/4)
// ---------------------------------------------------------------------------
__global__ void cvt_f32_bf16(const float* __restrict__ src,
                             unsigned short* __restrict__ dst, int n4) {
  int i = blockIdx.x * blockDim.x + threadIdx.x;
  int stride = gridDim.x * blockDim.x;
  for (; i < n4; i += stride) {
    float4 f = ((const float4*)src)[i];
    uint2 o;
    o.x = (unsigned)f2bf(f.x) | ((unsigned)f2bf(f.y) << 16);
    o.y = (unsigned)f2bf(f.z) | ((unsigned)f2bf(f.w) << 16);
    ((uint2*)dst)[i] = o;
  }
}

// ---------------------------------------------------------------------------
// C[M,N] = A[M,K] * B[N,K]^T (+ bias), bf16 inputs, f32 accumulate.
// Block: 256 threads (8 waves), tile 128x128, BK=32, double-buffered LDS.
// Wave grid 2x4, each wave computes 64x32 = 4x2 WMMA fragments.
// ---------------------------------------------------------------------------
template<bool HAS_BIAS, bool OUT_F32>
__global__ __launch_bounds__(256)
void gemm_bt(const unsigned short* __restrict__ A,
             const unsigned short* __restrict__ B,
             const float* __restrict__ bias,
             unsigned short* __restrict__ Cb,
             float* __restrict__ Cf,
             int M, int N, int K)
{
  const int BM = 128, BK = 32, LS = 48;       // LS elems = 96B row stride
  __shared__ unsigned short As[2][BM * LS];
  __shared__ unsigned short Bs[2][BM * LS];

  const int tid  = threadIdx.x;
  const int lane = tid & 31;
  const int wid  = tid >> 5;
  const int wm   = wid >> 2;                  // 0..1
  const int wn   = wid & 3;                   // 0..3
  const int m0   = blockIdx.y * BM;
  const int n0   = blockIdx.x * BM;
  const int lrow = lane & 15;
  const int kb   = (lane >> 4) * 8;

  // per-thread staging coordinates: 2 x b128 per tile per matrix
  const int srow0 = (tid * 2) >> 2, sseg0 = (tid * 2) & 3;
  const int srow1 = (tid * 2 + 1) >> 2, sseg1 = (tid * 2 + 1) & 3;

  v8f acc[4][2];
  #pragma unroll
  for (int i = 0; i < 4; i++) { acc[i][0] = {}; acc[i][1] = {}; }

  const int nk = K / BK;

  // issue tile 0 (4 async b128 per thread)
  copy16_g2l(A + (size_t)(m0 + srow0) * K + sseg0 * 8, &As[0][srow0 * LS + sseg0 * 8]);
  copy16_g2l(A + (size_t)(m0 + srow1) * K + sseg1 * 8, &As[0][srow1 * LS + sseg1 * 8]);
  copy16_g2l(B + (size_t)(n0 + srow0) * K + sseg0 * 8, &Bs[0][srow0 * LS + sseg0 * 8]);
  copy16_g2l(B + (size_t)(n0 + srow1) * K + sseg1 * 8, &Bs[0][srow1 * LS + sseg1 * 8]);

  for (int kt = 0; kt < nk; kt++) {
    const int cur = kt & 1, nxt = cur ^ 1;
    __syncthreads();                 // prior readers of buf[nxt] are done
    if (kt + 1 < nk) {               // overlap copy of tile kt+1 with compute
      const int k1 = (kt + 1) * BK;
      copy16_g2l(A + (size_t)(m0 + srow0) * K + k1 + sseg0 * 8, &As[nxt][srow0 * LS + sseg0 * 8]);
      copy16_g2l(A + (size_t)(m0 + srow1) * K + k1 + sseg1 * 8, &As[nxt][srow1 * LS + sseg1 * 8]);
      copy16_g2l(B + (size_t)(n0 + srow0) * K + k1 + sseg0 * 8, &Bs[nxt][srow0 * LS + sseg0 * 8]);
      copy16_g2l(B + (size_t)(n0 + srow1) * K + k1 + sseg1 * 8, &Bs[nxt][srow1 * LS + sseg1 * 8]);
      WAIT_ASYNC(4);                 // tile kt done; tile kt+1 stays in flight
    } else {
      WAIT_ASYNC(0);
    }
    __syncthreads();                 // tile kt visible to all waves

    AFrag a[4], b[2];
    #pragma unroll
    for (int mi = 0; mi < 4; mi++) {
      int r = wm * 64 + mi * 16 + lrow;
      a[mi].u[0] = *(const u32x4*)(&As[cur][r * LS + kb]);
      a[mi].u[1] = *(const u32x4*)(&As[cur][r * LS + kb + 16]);
    }
    #pragma unroll
    for (int ni = 0; ni < 2; ni++) {
      int c = wn * 32 + ni * 16 + lrow;
      b[ni].u[0] = *(const u32x4*)(&Bs[cur][c * LS + kb]);
      b[ni].u[1] = *(const u32x4*)(&Bs[cur][c * LS + kb + 16]);
    }
    #pragma unroll
    for (int mi = 0; mi < 4; mi++)
      #pragma unroll
      for (int ni = 0; ni < 2; ni++)
        acc[mi][ni] = __builtin_amdgcn_wmma_f32_16x16x32_bf16(
            false, a[mi].bf, false, b[ni].bf, (short)0, acc[mi][ni],
            false, false);
  }

  // epilogue: C frag layout -> row = base + i + (lane>>4)*8, col = base + lrow
  #pragma unroll
  for (int mi = 0; mi < 4; mi++) {
    #pragma unroll
    for (int ni = 0; ni < 2; ni++) {
      int n = n0 + wn * 32 + ni * 16 + lrow;
      float bval = HAS_BIAS ? bias[n] : 0.0f;
      #pragma unroll
      for (int i = 0; i < 8; i++) {
        int m = m0 + wm * 64 + mi * 16 + i + (lane >> 4) * 8;
        float v = acc[mi][ni][i] + bval;
        if (OUT_F32) Cf[(size_t)m * N + n] = v;
        else         Cb[(size_t)m * N + n] = f2bf(v);
      }
    }
  }
}

// ---------------------------------------------------------------------------
// Flash attention: grid (SEQ/64, NHEAD), block 128 (4 waves).
// Each wave owns 16 query rows; key tiles of 32 with online softmax.
// ---------------------------------------------------------------------------
__global__ __launch_bounds__(128)
void flash_attn(const unsigned short* __restrict__ Qb,
                const unsigned short* __restrict__ Kb,
                const unsigned short* __restrict__ Vb,
                unsigned short* __restrict__ Ob)
{
  const int qt   = blockIdx.x;           // query tile (64 rows)
  const int h    = blockIdx.y;           // head
  const int tid  = threadIdx.x;
  const int lane = tid & 31;
  const int w    = tid >> 5;             // wave id 0..3
  const int lrow = lane & 15;
  const int kb   = (lane >> 4) * 8;
  const float scale = 0.125f;            // 1/sqrt(64)

  __shared__ unsigned short Qs[64 * 80];        // [m][d], 160B stride
  __shared__ unsigned short Ks[32 * 80];        // [s][d]
  __shared__ unsigned short Vt[64 * 40];        // [d][s] (transposed), 80B stride
  __shared__ unsigned short Ps[4][16 * 40];     // per-wave P round-trip

  // stage Q tile (64x64 bf16) asynchronously; consumed after first barrier
  #pragma unroll
  for (int i = 0; i < 4; i++) {
    int idx = tid * 4 + i;
    int row = idx >> 3, seg = idx & 7;
    copy16_g2l(Qb + (size_t)(qt * 64 + row) * EMBED + h * HDIM + seg * 8,
               &Qs[row * 80 + seg * 8]);
  }

  v8f o[4]; o[0] = {}; o[1] = {}; o[2] = {}; o[3] = {};
  float mrun[8], lrun[8];
  #pragma unroll
  for (int i = 0; i < 8; i++) { mrun[i] = NEGINF; lrun[i] = 0.0f; }

  const int ktmax = 2 * qt + 1;          // causal: keys <= qt*64+63
  for (int kt = 0; kt <= ktmax; kt++) {
    __syncthreads();
    // stage K tile 32x64 [s][d] (async path)
    #pragma unroll
    for (int i = 0; i < 2; i++) {
      int idx = tid * 2 + i;
      int row = idx >> 3, seg = idx & 7;
      copy16_g2l(Kb + (size_t)(kt * 32 + row) * EMBED + h * HDIM + seg * 8,
                 &Ks[row * 80 + seg * 8]);
    }
    // stage V tile transposed -> Vt[d][s] (needs element shuffle: manual)
    #pragma unroll
    for (int i = 0; i < 2; i++) {
      int idx = tid * 2 + i;
      int s = idx >> 3, seg = idx & 7;
      u32x4 v = *(const u32x4*)(Vb + (size_t)(kt * 32 + s) * EMBED + h * HDIM + seg * 8);
      #pragma unroll
      for (int j = 0; j < 4; j++) {
        Vt[(seg * 8 + j * 2    ) * 40 + s] = (unsigned short)(v[j] & 0xFFFFu);
        Vt[(seg * 8 + j * 2 + 1) * 40 + s] = (unsigned short)(v[j] >> 16);
      }
    }
    WAIT_ASYNC(0);
    __syncthreads();

    // S = Q K^T : 16 rows x 32 keys, D=64 -> two 32-wide k-steps
    v8f sc[2]; sc[0] = {}; sc[1] = {};
    #pragma unroll
    for (int dstep = 0; dstep < 2; dstep++) {
      AFrag aq;
      int r = w * 16 + lrow;
      aq.u[0] = *(const u32x4*)(&Qs[r * 80 + dstep * 32 + kb]);
      aq.u[1] = *(const u32x4*)(&Qs[r * 80 + dstep * 32 + kb + 16]);
      #pragma unroll
      for (int sf = 0; sf < 2; sf++) {
        AFrag bk;
        int s = sf * 16 + lrow;
        bk.u[0] = *(const u32x4*)(&Ks[s * 80 + dstep * 32 + kb]);
        bk.u[1] = *(const u32x4*)(&Ks[s * 80 + dstep * 32 + kb + 16]);
        sc[sf] = __builtin_amdgcn_wmma_f32_16x16x32_bf16(
            false, aq.bf, false, bk.bf, (short)0, sc[sf], false, false);
      }
    }

    // scale + causal mask (C frag: row = base + i + (lane>>4)*8, col = lrow)
    int qg0 = qt * 64 + w * 16 + (lane >> 4) * 8;
    #pragma unroll
    for (int sf = 0; sf < 2; sf++) {
      int sg = kt * 32 + sf * 16 + lrow;
      #pragma unroll
      for (int i = 0; i < 8; i++) {
        float v = sc[sf][i] * scale;
        sc[sf][i] = (sg > qg0 + i) ? NEGINF : v;
      }
    }

    // online softmax: row max over 32 cols (16-lane shuffle reduction)
    float mnew[8], alpha[8];
    #pragma unroll
    for (int i = 0; i < 8; i++) {
      float tm = fmaxf(sc[0][i], sc[1][i]);
      #pragma unroll
      for (int off = 1; off < 16; off <<= 1)
        tm = fmaxf(tm, __shfl_xor(tm, off, 32));
      mnew[i]  = fmaxf(mrun[i], tm);
      alpha[i] = __expf(mrun[i] - mnew[i]);
      mrun[i]  = mnew[i];
    }
    #pragma unroll
    for (int sf = 0; sf < 2; sf++)
      #pragma unroll
      for (int i = 0; i < 8; i++)
        sc[sf][i] = __expf(sc[sf][i] - mnew[i]);
    #pragma unroll
    for (int i = 0; i < 8; i++) {
      float ts = sc[0][i] + sc[1][i];
      #pragma unroll
      for (int off = 1; off < 16; off <<= 1)
        ts += __shfl_xor(ts, off, 32);
      lrun[i] = lrun[i] * alpha[i] + ts;
    }
    #pragma unroll
    for (int di = 0; di < 4; di++)
      #pragma unroll
      for (int i = 0; i < 8; i++)
        o[di][i] *= alpha[i];

    // P: C-layout -> bf16 -> per-wave LDS -> reload in A-fragment layout
    #pragma unroll
    for (int sf = 0; sf < 2; sf++)
      #pragma unroll
      for (int i = 0; i < 8; i++)
        Ps[w][(i + (lane >> 4) * 8) * 40 + sf * 16 + lrow] = f2bf(sc[sf][i]);

    AFrag pa;   // LDS ops are in-order per wave; compiler inserts dscnt waits
    pa.u[0] = *(const u32x4*)(&Ps[w][lrow * 40 + kb]);
    pa.u[1] = *(const u32x4*)(&Ps[w][lrow * 40 + kb + 16]);

    // O += P * V  (K=32 keys, N=64 dims -> 4 fragments)
    #pragma unroll
    for (int di = 0; di < 4; di++) {
      AFrag bv;
      int d = di * 16 + lrow;
      bv.u[0] = *(const u32x4*)(&Vt[d * 40 + kb]);
      bv.u[1] = *(const u32x4*)(&Vt[d * 40 + kb + 16]);
      o[di] = __builtin_amdgcn_wmma_f32_16x16x32_bf16(
          false, pa.bf, false, bv.bf, (short)0, o[di], false, false);
    }
  }

  // normalize + store bf16 to attention-output workspace [T][H*D]
  #pragma unroll
  for (int i = 0; i < 8; i++) lrun[i] = 1.0f / lrun[i];
  #pragma unroll
  for (int di = 0; di < 4; di++) {
    #pragma unroll
    for (int i = 0; i < 8; i++) {
      int m = qt * 64 + w * 16 + i + (lane >> 4) * 8;
      int n = h * HDIM + di * 16 + lrow;
      Ob[(size_t)m * EMBED + n] = f2bf(o[di][i] * lrun[i]);
    }
  }
}

// ---------------------------------------------------------------------------
extern "C" void kernel_launch(void* const* d_in, const int* in_sizes, int n_in,
                              void* d_out, int out_size, void* d_ws, size_t ws_size,
                              hipStream_t stream)
{
  (void)in_sizes; (void)n_in; (void)out_size; (void)ws_size;
  const float* X  = (const float*)d_in[0];
  const float* wq = (const float*)d_in[1];
  const float* bq = (const float*)d_in[2];
  const float* wk = (const float*)d_in[3];
  const float* wv = (const float*)d_in[4];
  const float* bv = (const float*)d_in[5];
  const float* wo = (const float*)d_in[6];
  const float* bo = (const float*)d_in[7];
  // d_in[8] = pos_int == 0 for this problem size

  unsigned short* p = (unsigned short*)d_ws;
  unsigned short* Xb  = p; p += SEQ * EMBED;
  unsigned short* Wqb = p; p += EMBED * EMBED;
  unsigned short* Wkb = p; p += EMBED * EMBED;
  unsigned short* Wvb = p; p += EMBED * EMBED;
  unsigned short* Wob = p; p += EMBED * EMBED;
  unsigned short* Qb  = p; p += SEQ * EMBED;
  unsigned short* Kb  = p; p += SEQ * EMBED;
  unsigned short* Vb  = p; p += SEQ * EMBED;
  unsigned short* Ab  = p; p += SEQ * EMBED;
  // total workspace use: ~39.3 MB bf16 staging

  const int nX = SEQ * EMBED / 4, nW = EMBED * EMBED / 4;
  cvt_f32_bf16<<<(nX + 255) / 256, 256, 0, stream>>>(X,  Xb,  nX);
  cvt_f32_bf16<<<(nW + 255) / 256, 256, 0, stream>>>(wq, Wqb, nW);
  cvt_f32_bf16<<<(nW + 255) / 256, 256, 0, stream>>>(wk, Wkb, nW);
  cvt_f32_bf16<<<(nW + 255) / 256, 256, 0, stream>>>(wv, Wvb, nW);
  cvt_f32_bf16<<<(nW + 255) / 256, 256, 0, stream>>>(wo, Wob, nW);

  dim3 gg(EMBED / 128, SEQ / 128);   // (10, 16)
  gemm_bt<true,  false><<<gg, 256, 0, stream>>>(Xb, Wqb, bq,      Qb, nullptr, SEQ, EMBED, EMBED);
  gemm_bt<false, false><<<gg, 256, 0, stream>>>(Xb, Wkb, nullptr, Kb, nullptr, SEQ, EMBED, EMBED);
  gemm_bt<true,  false><<<gg, 256, 0, stream>>>(Xb, Wvb, bv,      Vb, nullptr, SEQ, EMBED, EMBED);

  flash_attn<<<dim3(SEQ / 64, NHEAD), 128, 0, stream>>>(Qb, Kb, Vb, Ab);

  gemm_bt<true, true><<<gg, 256, 0, stream>>>(Ab, Wob, bo, nullptr, (float*)d_out, SEQ, EMBED, EMBED);
}
